// CutlassLinear_84748294685034
// MI455X (gfx1250) — matrix operand order repaired and checked by reading, not compiled, not run on hardware
//
#include <hip/hip_runtime.h>

typedef __attribute__((ext_vector_type(16))) _Float16 v16h;
typedef __attribute__((ext_vector_type(8)))  _Float16 v8h;
typedef __attribute__((ext_vector_type(4)))  _Float16 v4h;
typedef __attribute__((ext_vector_type(8)))  float    v8f;

#define M_DIM 4096
#define N_DIM 4096
#define K_DIM 4096

#define TM 128              // block tile M (tokens)
#define TN 128              // block tile N (out features)
#define KB 64               // K chunk staged through LDS (2 WMMA k-steps)
#define LDT 72              // LDS row stride in halves (144B: conflict-free)
#define NSTAGES (K_DIM / KB)

static __device__ __forceinline__ v16h concat16(v8h lo, v8h hi) {
    return __builtin_shufflevector(lo, hi, 0, 1, 2, 3, 4, 5, 6, 7,
                                           8, 9, 10, 11, 12, 13, 14, 15);
}

// ---------------------------------------------------------------------------
// Phase 0: zero the absmax slot (ws poisoned 0xAA; re-init every call).
// ---------------------------------------------------------------------------
__global__ void init_absmax_kernel(unsigned int* wsMax) {
    if (threadIdx.x == 0) wsMax[0] = 0u;   // bit pattern of +0.0f
}

// ---------------------------------------------------------------------------
// Phase 1: max|W| over 16M floats. float4 grid-stride + LDS tree + int-typed
// atomic max (bit order == float order for non-negative floats).
// ---------------------------------------------------------------------------
__global__ __launch_bounds__(256) void absmax_kernel(const float* __restrict__ W,
                                                     unsigned int* wsMax) {
    const int tid    = blockIdx.x * blockDim.x + threadIdx.x;
    const int stride = gridDim.x * blockDim.x;
    const float4* W4 = (const float4*)W;
    const int n4 = (K_DIM * N_DIM) / 4;

    float m = 0.0f;
    for (int i = tid; i < n4; i += stride) {
        float4 v = W4[i];
        m = fmaxf(m, fmaxf(fmaxf(fabsf(v.x), fabsf(v.y)),
                           fmaxf(fabsf(v.z), fabsf(v.w))));
    }

    __shared__ float red[256];
    red[threadIdx.x] = m;
    __syncthreads();
    for (int s = 128; s > 0; s >>= 1) {
        if (threadIdx.x < s)
            red[threadIdx.x] = fmaxf(red[threadIdx.x], red[threadIdx.x + s]);
        __syncthreads();
    }
    if (threadIdx.x == 0)
        atomicMax((int*)wsMax, __float_as_int(red[0]));
}

// ---------------------------------------------------------------------------
// Phase 2: software-pipelined WMMA GEMM.
//   y[t,o] = (sum_k x_f16[t,k] * q[o,k]) * (max/127) + bias[o]
// q = trunc(W * 127/max) kept as exact integer values in f16 (|q| <= 127
// is exactly representable; no clamp needed since the global scale bounds
// |W*qScale| by 127). f32 accumulation via v_wmma_f32_16x16x32_f16.
//
// Pipeline: while stage s is computed from LDS, stage s+1's 16
// global_load_b128 are already in flight into a register staging set; after
// the read-barrier they are converted/quantized and ds_stored for the next
// iteration. Global (L2) latency is hidden behind 16 WMMAs per wave.
// ---------------------------------------------------------------------------
__global__ __launch_bounds__(256) void qlinear_wmma_kernel(
        const float* __restrict__ X,      // [M, K] row-major
        const float* __restrict__ W,      // [N, K] row-major
        const float* __restrict__ bias,   // [N]
        float* __restrict__ Y,            // [M, N] row-major
        const float* __restrict__ wsMax) {

    __shared__ _Float16 ldsA[TM * LDT];   // x tile, f16
    __shared__ _Float16 ldsB[TN * LDT];   // quantized w tile, f16 (integers)

    const float maxAbs = wsMax[0];
    const float qScale = 127.0f / maxAbs;
    const float dScale = maxAbs / 127.0f;

    const int rowBase = blockIdx.y * TM;  // token rows
    const int colBase = blockIdx.x * TN;  // output features

    const int tid  = (int)threadIdx.x;
    const int lane = tid & 31;
    const int wave = tid >> 5;
    const int m0 = (wave & 3) * 32;       // wave sub-tile: 32 rows
    const int n0 = (wave >> 2) * 64;      // wave sub-tile: 64 cols

    const int fr = lane & 15;             // fragment row/col within 16
    const int kh = lane >> 4;             // K half-select

    // Per-thread tile-copy coordinates (8 float4 per tile per thread).
    const int cr  = tid >> 4;             // base row  0..15 (rows cr, cr+16, ...)
    const int cc4 = tid & 15;             // float4 column 0..15

    v8f acc[2][4];
#pragma unroll
    for (int mt = 0; mt < 2; ++mt)
#pragma unroll
        for (int nt = 0; nt < 4; ++nt)
            acc[mt][nt] = (v8f){0.f, 0.f, 0.f, 0.f, 0.f, 0.f, 0.f, 0.f};

    float4 xs[8], ws[8];                  // register staging for next stage

    // ---- prologue: fetch + stage K-chunk 0 ----
#pragma unroll
    for (int i = 0; i < 8; ++i) {
        const int r = cr + i * 16;
        xs[i] = *(const float4*)&X[(size_t)(rowBase + r) * K_DIM + cc4 * 4];
        ws[i] = *(const float4*)&W[(size_t)(colBase + r) * K_DIM + cc4 * 4];
    }
#pragma unroll
    for (int i = 0; i < 8; ++i) {
        const int r = cr + i * 16;
        v4h xh;
        xh[0] = (_Float16)xs[i].x;  xh[1] = (_Float16)xs[i].y;
        xh[2] = (_Float16)xs[i].z;  xh[3] = (_Float16)xs[i].w;
        *(v4h*)&ldsA[r * LDT + cc4 * 4] = xh;
        v4h qh;
        qh[0] = (_Float16)truncf(ws[i].x * qScale);
        qh[1] = (_Float16)truncf(ws[i].y * qScale);
        qh[2] = (_Float16)truncf(ws[i].z * qScale);
        qh[3] = (_Float16)truncf(ws[i].w * qScale);
        *(v4h*)&ldsB[r * LDT + cc4 * 4] = qh;
    }
    __syncthreads();

#pragma unroll 1
    for (int s = 0; s < NSTAGES; ++s) {
        const bool more = (s + 1) < NSTAGES;

        // ---- issue next stage's global loads (overlap with WMMA below) ----
        if (more) {
            const int kb = (s + 1) * KB;
#pragma unroll
            for (int i = 0; i < 8; ++i) {
                const int r = cr + i * 16;
                xs[i] = *(const float4*)&X[(size_t)(rowBase + r) * K_DIM + kb + cc4 * 4];
                ws[i] = *(const float4*)&W[(size_t)(colBase + r) * K_DIM + kb + cc4 * 4];
            }
        }

        // ---- compute current stage: two 16x16x32 WMMA k-steps ----
#pragma unroll
        for (int ks = 0; ks < KB; ks += 32) {
            // A fragments (16x32 f16): lane holds row m, K = kh*8+{0..7} and 16+kh*8+{0..7}
            v16h a[2];
#pragma unroll
            for (int mt = 0; mt < 2; ++mt) {
                const _Float16* p = &ldsA[(m0 + mt * 16 + fr) * LDT + ks + kh * 8];
                a[mt] = concat16(*(const v8h*)p, *(const v8h*)(p + 16));
            }
            // B fragments (32x16 f16): lane holds col n, K = kh*8+{0..7} and 16+kh*8+{0..7}
            v16h b[4];
#pragma unroll
            for (int nt = 0; nt < 4; ++nt) {
                const _Float16* p = &ldsB[(n0 + nt * 16 + fr) * LDT + ks + kh * 8];
                b[nt] = concat16(*(const v8h*)p, *(const v8h*)(p + 16));
            }
#pragma unroll
            for (int mt = 0; mt < 2; ++mt)
#pragma unroll
                for (int nt = 0; nt < 4; ++nt)
                    acc[mt][nt] = __builtin_amdgcn_wmma_f32_16x16x32_f16(
                        /*neg_a=*/false, a[mt], /*neg_b=*/false, b[nt],
                        /*c_mod=*/(short)0, acc[mt][nt],
                        /*reuse_a=*/false, /*reuse_b=*/false);
        }

        __syncthreads();                  // all waves done reading this stage

        // ---- convert + stage next chunk into LDS ----
        if (more) {
#pragma unroll
            for (int i = 0; i < 8; ++i) {
                const int r = cr + i * 16;
                v4h xh;
                xh[0] = (_Float16)xs[i].x;  xh[1] = (_Float16)xs[i].y;
                xh[2] = (_Float16)xs[i].z;  xh[3] = (_Float16)xs[i].w;
                *(v4h*)&ldsA[r * LDT + cc4 * 4] = xh;
                v4h qh;
                qh[0] = (_Float16)truncf(ws[i].x * qScale);
                qh[1] = (_Float16)truncf(ws[i].y * qScale);
                qh[2] = (_Float16)truncf(ws[i].z * qScale);
                qh[3] = (_Float16)truncf(ws[i].w * qScale);
                *(v4h*)&ldsB[r * LDT + cc4 * 4] = qh;
            }
            __syncthreads();              // next stage visible to all waves
        }
    }

    // ---- epilogue: dequant scale + bias, write fp32 ----
    // C layout: VGPR r, lane -> M = r + (lane>>4)*8, N = lane&15
#pragma unroll
    for (int nt = 0; nt < 4; ++nt) {
        const int o   = colBase + n0 + nt * 16 + fr;
        const float bv = bias[o];
#pragma unroll
        for (int mt = 0; mt < 2; ++mt) {
#pragma unroll
            for (int r = 0; r < 8; ++r) {
                const int t = rowBase + m0 + mt * 16 + r + kh * 8;
                Y[(size_t)t * N_DIM + o] = acc[mt][nt][r] * dScale + bv;
            }
        }
    }
}

// ---------------------------------------------------------------------------
extern "C" void kernel_launch(void* const* d_in, const int* in_sizes, int n_in,
                              void* d_out, int out_size, void* d_ws, size_t ws_size,
                              hipStream_t stream) {
    const float* x    = (const float*)d_in[0];   // [4096, 4096]
    const float* w    = (const float*)d_in[1];   // [4096, 4096]
    const float* bias = (const float*)d_in[2];   // [4096]
    float* out = (float*)d_out;                  // [4096, 4096]
    unsigned int* wsMax = (unsigned int*)d_ws;   // 4 bytes used

    init_absmax_kernel<<<1, 32, 0, stream>>>(wsMax);
    absmax_kernel<<<512, 256, 0, stream>>>(w, wsMax);

    dim3 grid(N_DIM / TN, M_DIM / TM);           // 32 x 32 blocks
    qlinear_wmma_kernel<<<grid, 256, 0, stream>>>(x, w, bias, out,
                                                  (const float*)wsMax);
}
